// KernelXML_48765058679048
// MI455X (gfx1250) — matrix operand order repaired and checked by reading, compiled, not run on hardware
//
#include <hip/hip_runtime.h>
#include <hip/hip_bf16.h>

// ---------------- model dims ----------------
#define DIM      256      // EMBED_DIM
#define HEADS    4
#define DH       64
#define RQ       2
#define MF       32
#define BATCH    4096
#define SEQ      64
#define NLABELS  262144
#define NPOS     20480

typedef __attribute__((ext_vector_type(16))) _Float16 v16h;
typedef __attribute__((ext_vector_type(8)))  float    v8f;

// Gauss-Laguerre (R=2) nodes/weights divided by C = 2 + 1e-6
__device__ __constant__ const float kS[2]  = {0.2928930723688f, 1.7071059276134f};   // nodes / C
__device__ __constant__ const float kW[2]  = {0.4267764819094f, 0.0732232681003f};   // weights / C

// ---------------- zero init (atomics must start from 0 every call) ------------
__global__ void zero_kernel(float* __restrict__ Z, float* __restrict__ outp) {
  if (threadIdx.x < DIM) Z[threadIdx.x] = 0.0f;
  if (threadIdx.x == 0)  outp[0] = 0.0f;
}

// ---------------- one-time omega -> WMMA B-fragment swizzle -------------------
// Layout: bw[((h*4+ct)*2+kb) * 32 + lane] is one v16h (32B) fragment slice,
// in the contiguous k order shared with the A side (K = kb*32 + half*16 + e,
// N = lane&15). Consumers reload with 2 coalesced b128 per fragment.
__global__ __launch_bounds__(32)
void prep_bfrag(const float* __restrict__ omega, _Float16* __restrict__ bw) {
  const int lane = threadIdx.x, half = lane >> 4, lml = lane & 15;
#pragma unroll
  for (int h = 0; h < HEADS; ++h)
#pragma unroll
    for (int ct = 0; ct < 4; ++ct) {
      const int col = ct * 16 + lml;          // f' = r*32 + m
      const int r = col >> 5, m = col & 31;
#pragma unroll
      for (int kb = 0; kb < 2; ++kb) {
        const int f = (h * 4 + ct) * 2 + kb;
#pragma unroll
        for (int e = 0; e < 16; ++e) {
          const int k = kb * 32 + half * 16 + e;
          bw[(f * 32 + lane) * 16 + e] = (_Float16)omega[((r * HEADS + h) * DH + k) * MF + m];
        }
      }
    }
}

// ---------------- query = masked mean of embedding gather ---------------------
__global__ __launch_bounds__(256)
void query_kernel(const int* __restrict__ indices, const float* __restrict__ mask,
                  const float* __restrict__ table, float* __restrict__ query) {
  __shared__ int   sidx[SEQ];
  __shared__ float smask[SEQ];
  __shared__ float smsum;
  const int b = blockIdx.x, d = threadIdx.x;
  if (d < SEQ) { sidx[d] = indices[b * SEQ + d]; smask[d] = mask[b * SEQ + d]; }
  __syncthreads();
  if (d == 0) {
    float ms = 0.0f;
    for (int s = 0; s < SEQ; ++s) ms += smask[s];
    smsum = ms;
  }
  __syncthreads();
  float acc = 0.0f;
  for (int s = 0; s < SEQ; ++s)
    acc += table[sidx[s] * DIM + d] * smask[s];     // 100000*256 < 2^31: int math
  query[b * DIM + d] = acc / fmaxf(smsum, 1e-9f);
}

// ---------------- SLAY feature kernel (WMMA core) -----------------------------
// MODE 0: store phi rows to out[N,256] (optional row gather via gidx)
// MODE 1: accumulate column sums into out[256] (Z_vec) with atomics
template <int MODE>
__global__ __launch_bounds__(32)
void phi_kernel(const float* __restrict__ X, const int* __restrict__ gidx,
                int ntiles, const v16h* __restrict__ bW, float* __restrict__ out) {
  __shared__ _Float16 xn[16][264];          // normalized rows, f16, padded

  const int lane = threadIdx.x;             // wave32
  const int half = lane >> 4;               // 0/1
  const int lml  = lane & 15;

  // --- B fragments: pre-swizzled, 2 coalesced b128 loads each, register-resident
  v16h bfrag[HEADS][4][2];
#pragma unroll
  for (int h = 0; h < HEADS; ++h)
#pragma unroll
    for (int ct = 0; ct < 4; ++ct)
#pragma unroll
      for (int kb = 0; kb < 2; ++kb)
        bfrag[h][ct][kb] = bW[((h * 4 + ct) * 2 + kb) * 32 + lane];

  // per-r feature constants (constant-folded)
  const float sq2s[2] = {sqrtf(2.0f * kS[0]), sqrtf(2.0f * kS[1])};
  const float coef[2] = {(1.0f / (float)MF) * sqrtf(kW[0]),
                         (1.0f / (float)MF) * sqrtf(kW[1])};

  float zacc[HEADS][4];
  if (MODE == 1) {
#pragma unroll
    for (int h = 0; h < HEADS; ++h)
#pragma unroll
      for (int ct = 0; ct < 4; ++ct) zacc[h][ct] = 0.0f;
  }

  for (int tile = blockIdx.x; tile < ntiles; tile += gridDim.x) {
    // --- load 16 rows, per-head L2 normalize, write f16 to LDS ---
    {
      const int row  = lml;
      const int rowg = gidx ? gidx[tile * 16 + row] : (tile * 16 + row);
      const float4* src = (const float4*)(X + rowg * DIM) + half * 32;  // 2 heads/lane
#pragma unroll
      for (int hh = 0; hh < 2; ++hh) {
        const float4* hs = src + hh * 16;
        float4 vv[16];
        float ss = 0.0f;
#pragma unroll
        for (int i = 0; i < 16; ++i) {
          vv[i] = hs[i];
          ss += vv[i].x * vv[i].x + vv[i].y * vv[i].y +
                vv[i].z * vv[i].z + vv[i].w * vv[i].w;
        }
        const float inv = 1.0f / fmaxf(sqrtf(ss), 1e-6f);
        const int hcol = half * 128 + hh * DH;
#pragma unroll
        for (int i = 0; i < 16; ++i) {
          xn[row][hcol + 4 * i + 0] = (_Float16)(vv[i].x * inv);
          xn[row][hcol + 4 * i + 1] = (_Float16)(vv[i].y * inv);
          xn[row][hcol + 4 * i + 2] = (_Float16)(vv[i].z * inv);
          xn[row][hcol + 4 * i + 3] = (_Float16)(vv[i].w * inv);
        }
      }
    }
    __syncthreads();

    // single store base per tile: all output offsets become immediates
    float* op = (MODE == 0) ? (out + (tile * 16 + half * 8) * DIM + lml) : out;

#pragma unroll
    for (int h = 0; h < HEADS; ++h) {
      // A fragments (contiguous k order, same as B), K = d within head
      v16h a0, a1;
#pragma unroll
      for (int kb = 0; kb < 2; ++kb) {
        v16h a;
#pragma unroll
        for (int e = 0; e < 16; ++e) {
          const int g = e >> 3, o = e & 7;
          const int k = kb * 32 + half * 16 + e;
          a[e] = xn[lml][h * DH + k];
        }
        if (kb == 0) a0 = a; else a1 = a;
      }
#pragma unroll
      for (int ct = 0; ct < 4; ++ct) {
        v8f c = {};
        c = __builtin_amdgcn_wmma_f32_16x16x32_f16(false, a0, false, bfrag[h][ct][0],
                                                   (short)0, c, false, false);
        c = __builtin_amdgcn_wmma_f32_16x16x32_f16(false, a1, false, bfrag[h][ct][1],
                                                   (short)0, c, false, false);
        const int r = ct >> 1;                       // f' = ct*16+n, r = f'>>5
        const float s = kS[r], q2 = sq2s[r], cf = coef[r];
        if (MODE == 0) {
          const int oc = h * 64 + ct * 16;           // compile-time constant
#pragma unroll
          for (int v = 0; v < 8; ++v) {              // C: M = v + 8*half
            const float p = c[v];
            const float t = fminf(fmaxf(p * q2 - s, -10.0f), 10.0f);
            op[oc + v * DIM] = p * p * cf * __expf(t);
          }
        } else {
          float acc = 0.0f;
#pragma unroll
          for (int v = 0; v < 8; ++v) {
            const float p = c[v];
            const float t = fminf(fmaxf(p * q2 - s, -10.0f), 10.0f);
            acc += p * p * cf * __expf(t);
          }
          zacc[h][ct] += acc;
        }
      }
    }
    __syncthreads();
  }

  if (MODE == 1) {
#pragma unroll
    for (int h = 0; h < HEADS; ++h)
#pragma unroll
      for (int ct = 0; ct < 4; ++ct) {
        float z = zacc[h][ct];
        z += __shfl_xor(z, 16, 32);                  // both halves share a column
        if (half == 0) atomicAdd(&out[h * 64 + ct * 16 + lml], z);
      }
  }
}

// ---------------- log_Z[b] = log(phi_q[b] . Z + 1e-6) -------------------------
__global__ __launch_bounds__(256)
void logz_kernel(const float* __restrict__ phi_q, const float* __restrict__ Z,
                 float* __restrict__ logZ) {
  const int wave = threadIdx.x >> 5, lane = threadIdx.x & 31;
  const int b = blockIdx.x * 8 + wave;
  float acc = 0.0f;
#pragma unroll
  for (int i = 0; i < 8; ++i)
    acc += phi_q[b * DIM + lane + 32 * i] * Z[lane + 32 * i];
#pragma unroll
  for (int o = 16; o; o >>= 1) acc += __shfl_xor(acc, o, 32);
  if (lane == 0) logZ[b] = __logf(acc + 1e-6f);
}

// ---------------- loss = -sum(log(num) - logZ[sid]) / B -----------------------
__global__ __launch_bounds__(256)
void loss_kernel(const float* __restrict__ phi_l, const float* __restrict__ phi_q,
                 const int* __restrict__ sid, const float* __restrict__ logZ,
                 float* __restrict__ outp) {
  __shared__ float bsum;
  if (threadIdx.x == 0) bsum = 0.0f;
  __syncthreads();
  const int wave = threadIdx.x >> 5, lane = threadIdx.x & 31;
  const int j = blockIdx.x * 8 + wave;               // grid sized exactly
  const int b = sid[j];
  float acc = 0.0f;
#pragma unroll
  for (int i = 0; i < 8; ++i)
    acc += phi_l[j * DIM + lane + 32 * i] * phi_q[b * DIM + lane + 32 * i];
#pragma unroll
  for (int o = 16; o; o >>= 1) acc += __shfl_xor(acc, o, 32);
  if (lane == 0) atomicAdd(&bsum, -(__logf(acc + 1e-6f) - logZ[b]));
  __syncthreads();
  if (threadIdx.x == 0) atomicAdd(outp, bsum * (1.0f / (float)BATCH));
}

// ---------------- launch ------------------------------------------------------
extern "C" void kernel_launch(void* const* d_in, const int* in_sizes, int n_in,
                              void* d_out, int out_size, void* d_ws, size_t ws_size,
                              hipStream_t stream) {
  const int*   indices     = (const int*)d_in[0];
  const float* mask        = (const float*)d_in[1];
  const int*   flat_labels = (const int*)d_in[2];
  const int*   sample_ids  = (const int*)d_in[3];
  const float* embed_table = (const float*)d_in[4];
  const float* W_vecs      = (const float*)d_in[5];
  const float* omega       = (const float*)d_in[6];
  float* out = (float*)d_out;

  float* ws    = (float*)d_ws;
  float* bWf   = ws;                                  // 32KB swizzled omega frags
  float* query = bWf + 8192;                          // 4096*256
  float* phi_q = query + (size_t)BATCH * DIM;         // 4096*256
  float* phi_l = phi_q + (size_t)BATCH * DIM;         // 20480*256
  float* Zv    = phi_l + (size_t)NPOS * DIM;          // 256
  float* logZ  = Zv + DIM;                            // 4096

  zero_kernel<<<1, 256, 0, stream>>>(Zv, out);
  prep_bfrag<<<1, 32, 0, stream>>>(omega, (_Float16*)bWf);
  query_kernel<<<BATCH, 256, 0, stream>>>(indices, mask, embed_table, query);
  // phi_q: 4096 rows = 256 tiles
  phi_kernel<0><<<256, 32, 0, stream>>>(query, nullptr, BATCH / 16,
                                        (const v16h*)bWf, phi_q);
  // Z_vec: stream all 262144 labels (16384 tiles, 8 per wave)
  phi_kernel<1><<<2048, 32, 0, stream>>>(W_vecs, nullptr, NLABELS / 16,
                                         (const v16h*)bWf, Zv);
  // phi of gathered positive labels: 20480 rows = 1280 tiles
  phi_kernel<0><<<1280, 32, 0, stream>>>(W_vecs, flat_labels, NPOS / 16,
                                         (const v16h*)bWf, phi_l);
  logz_kernel<<<BATCH / 8, 256, 0, stream>>>(phi_q, Zv, logZ);
  loss_kernel<<<NPOS / 8, 256, 0, stream>>>(phi_l, phi_q, sample_ids, logZ, out);
}
